// mLSTMCell_37374805409863
// MI455X (gfx1250) — matrix operand order repaired and checked by reading, not compiled, not run on hardware
//
#include <hip/hip_runtime.h>
#include <math.h>

typedef __attribute__((ext_vector_type(16))) _Float16 v16h;
typedef __attribute__((ext_vector_type(8)))  float    v8f;

#define B_   8
#define T_   2048
#define IN_  1024
#define HID_ 1024
#define H_   8
#define D_   64
#define HD_  512
#define NTOK (B_ * T_)   // 16384
#define L_   64          // chunk length
#define NC   (T_ / L_)   // 32 chunks
#define NCAT 2048        // q|k|v|o concatenated columns
#define LDA  40          // LDS row stride (halves): 32 data + 8 pad

// ---------------------------------------------------------------------------
// WMMA helpers (wave32, 16x16x32 f16 -> f32). A/B use contiguous k order: k = 16*(lane/16) + e.
// ---------------------------------------------------------------------------
__device__ inline v16h frag_A(const _Float16* p, int ld, int m0, int k0, int lane) {
  v16h a;
  int m  = m0 + (lane & 15);
  int kb = k0 + ((lane & 16) ? 16 : 0);
#pragma unroll
  for (int e = 0; e < 16; ++e) {
    int k = kb + e;
    a[e] = p[m * ld + k];
  }
  return a;
}
// B(k,n) from an n-major buffer: value = p[n*ld + k]
__device__ inline v16h frag_B_nmajor(const _Float16* p, int ld, int n0, int k0, int lane) {
  v16h b;
  int n  = n0 + (lane & 15);
  int kb = k0 + ((lane & 16) ? 16 : 0);
#pragma unroll
  for (int e = 0; e < 16; ++e) b[e] = p[n * ld + kb + e];
  return b;
}
// B(k,n) from a k-major buffer: value = p[k*ld + n]
__device__ inline v16h frag_B_kmajor(const _Float16* p, int ld, int k0, int n0, int lane) {
  v16h b;
  int n  = n0 + (lane & 15);
  int kb = k0 + ((lane & 16) ? 16 : 0);
#pragma unroll
  for (int e = 0; e < 16; ++e) b[e] = p[(kb + e) * ld + n];
  return b;
}
__device__ inline v8f wmma_f16(v16h a, v16h b, v8f c) {
  return __builtin_amdgcn_wmma_f32_16x16x32_f16(false, a, false, b, (short)0, c,
                                                false, false);
}

// ---------------------------------------------------------------------------
// Tensor Data Mover staging of a 64-row x 32-half tile into LDS with 8-half
// row padding (pad 4 DWORDs every 16 DWORDs => LDS stride = LDA halves).
// D# layout per CDNA5 ISA ch.8.3/8.4. One wave issues; TENSORcnt tracks it.
// This toolchain's builtin is the 6-arg form:
//   (u32x4 g0, i32x8 g1, i32x4 g2, i32x4 g3, i32x8 pad, i32 cpol)
// ---------------------------------------------------------------------------
#if __has_builtin(__builtin_amdgcn_tensor_load_to_lds) && \
    __has_builtin(__builtin_amdgcn_s_wait_tensorcnt)
#define HAVE_TDM 1
typedef __attribute__((ext_vector_type(4))) unsigned int u32x4;
typedef __attribute__((ext_vector_type(8))) int i32x8;
typedef __attribute__((ext_vector_type(4))) int i32x4;

__device__ inline void tdm_load_tile_f16(const _Float16* gsrc, void* lds_dst,
                                         int row_stride_elems) {
  // Generic pointers into LDS carry the LDS byte offset in addr[31:0]
  // (ISA 10.2 aperture mapping).
  unsigned lds_addr = (unsigned)(uintptr_t)lds_dst;
  unsigned long long ga = (unsigned long long)(uintptr_t)gsrc;
  u32x4 g0;
  g0[0] = 1u;                                    // count=1, user descriptor
  g0[1] = lds_addr;                              // lds_addr [63:32]
  g0[2] = (unsigned)ga;                          // global_addr lo
  g0[3] = ((unsigned)(ga >> 32) & 0x01FFFFFFu) | (2u << 30);  // addr hi | type=2
  i32x8 g1;
  // data_size=2B (code 1), pad_enable=1, pad_interval=16dw (code 3),
  // pad_amount=4dw (code 3), workgroup_mask=0 (not in cluster)
  g1[0] = (int)((1u << 16) | (1u << 20) | (3u << 22) | (3u << 25));
  unsigned td0 = (unsigned)row_stride_elems;     // tensor_dim0 = row length
  unsigned td1 = 0x000FFFFFu;                    // tensor_dim1: generous (in-range)
  g1[1] = (int)((td0 & 0xFFFFu) << 16);          // tensor_dim0[15:0] @ [63:48]
  g1[2] = (int)((td0 >> 16) | ((td1 & 0xFFFFu) << 16));  // dim0 hi | dim1 lo
  g1[3] = (int)((td1 >> 16) | (32u << 16));      // dim1 hi | tile_dim0=32
  g1[4] = 64;                                    // tile_dim1=64, tile_dim2=0
  g1[5] = row_stride_elems;                      // tensor_dim0_stride lo32
  g1[6] = 0;                                     // stride0 hi16 | stride1 lo16
  g1[7] = 0;
  i32x4 z4 = {0, 0, 0, 0};
  i32x8 z8 = {0, 0, 0, 0, 0, 0, 0, 0};
  __builtin_amdgcn_tensor_load_to_lds(g0, g1, z4, z4, z8, 0);
}
#else
#define HAVE_TDM 0
#endif

// Stage one A-tile and one B-tile (each 64x32 f16, global row stride `ld`).
__device__ inline void stage_pair(const _Float16* ga, const _Float16* gb,
                                  _Float16* la, _Float16* lb, int ld, int tid,
                                  int nthr) {
#if HAVE_TDM
  if (tid == 0) {
    tdm_load_tile_f16(ga, la, ld);
    tdm_load_tile_f16(gb, lb, ld);
  }
#else
  for (int i = tid; i < 64 * 32; i += nthr) {
    int r = i >> 5, c = i & 31;
    la[r * LDA + c] = ga[(size_t)r * ld + c];
    lb[r * LDA + c] = gb[(size_t)r * ld + c];
  }
#endif
}
__device__ inline void stage_wait(int tid) {
#if HAVE_TDM
  if (tid == 0) __builtin_amdgcn_s_wait_tensorcnt(0);
#endif
  __syncthreads();
}

// ---------------------------------------------------------------------------
// K0a: cast weights to f16. Wcat = [Wq ; Wk/8 ; Wv ; Wo], Wouth = W_out.
// ---------------------------------------------------------------------------
__global__ __launch_bounds__(256) void k_cast(
    const float* __restrict__ Wq, const float* __restrict__ Wk,
    const float* __restrict__ Wv, const float* __restrict__ Wo,
    const float* __restrict__ Wout, _Float16* __restrict__ Wcat,
    _Float16* __restrict__ Wouth) {
  int i = blockIdx.x * 256 + threadIdx.x;
  if (i < NCAT * IN_) {
    int row = i >> 10, col = i & 1023;
    int seg = row >> 9, r = row & 511;
    float v;
    if (seg == 0)      v = Wq[r * IN_ + col];
    else if (seg == 1) v = Wk[r * IN_ + col] * 0.125f;  // fold 1/sqrt(D)
    else if (seg == 2) v = Wv[r * IN_ + col];
    else               v = Wo[r * IN_ + col];
    Wcat[i] = (_Float16)v;
  } else {
    int j = i - NCAT * IN_;
    if (j < HID_ * HD_) Wouth[j] = (_Float16)Wout[j];
  }
}

// K0b: cast activations x -> f16 (so TDM can stage them without conversion).
__global__ __launch_bounds__(256) void k_castx(const float* __restrict__ x,
                                               _Float16* __restrict__ xh) {
  size_t i = ((size_t)blockIdx.x * 256 + threadIdx.x) * 4;
  float4 v = *(const float4*)(x + i);
  xh[i + 0] = (_Float16)v.x;
  xh[i + 1] = (_Float16)v.y;
  xh[i + 2] = (_Float16)v.z;
  xh[i + 3] = (_Float16)v.w;
}

// ---------------------------------------------------------------------------
// K1: fused projection GEMM  P = x @ Wcat^T  (16384 x 2048), f32 accum,
//     TDM double-buffered LDS staging. Epilogue: q|k|v|sigmoid(o).
// ---------------------------------------------------------------------------
__global__ __launch_bounds__(128) void k_proj(
    const _Float16* __restrict__ xh, const _Float16* __restrict__ Wcat,
    const float* __restrict__ bo, _Float16* __restrict__ qh,
    _Float16* __restrict__ kh, _Float16* __restrict__ vh,
    _Float16* __restrict__ oh) {
  __shared__ _Float16 As[2][64 * LDA];
  __shared__ _Float16 Bs[2][64 * LDA];
  int tid = threadIdx.x, lane = tid & 31, w = tid >> 5;
  int m0 = blockIdx.y * 64, n0 = blockIdx.x * 64;
  int wr = (w >> 1) * 32, wc = (w & 1) * 32;
  const _Float16* Ab = xh + (size_t)m0 * IN_;
  const _Float16* Bb = Wcat + (size_t)n0 * IN_;
  v8f acc[2][2] = {};
  stage_pair(Ab, Bb, As[0], Bs[0], IN_, tid, 128);
  stage_wait(tid);
  int p = 0;
  for (int k0 = 0; k0 < IN_; k0 += 32) {
    if (k0 + 32 < IN_)
      stage_pair(Ab + k0 + 32, Bb + k0 + 32, As[p ^ 1], Bs[p ^ 1], IN_, tid, 128);
    v16h a0 = frag_A(As[p], LDA, wr, 0, lane);
    v16h a1 = frag_A(As[p], LDA, wr + 16, 0, lane);
    v16h b0 = frag_B_nmajor(Bs[p], LDA, wc, 0, lane);
    v16h b1 = frag_B_nmajor(Bs[p], LDA, wc + 16, 0, lane);
    acc[0][0] = wmma_f16(a0, b0, acc[0][0]);
    acc[0][1] = wmma_f16(a0, b1, acc[0][1]);
    acc[1][0] = wmma_f16(a1, b0, acc[1][0]);
    acc[1][1] = wmma_f16(a1, b1, acc[1][1]);
    stage_wait(tid);
    p ^= 1;
  }
#pragma unroll
  for (int tm = 0; tm < 2; ++tm)
#pragma unroll
    for (int tn = 0; tn < 2; ++tn) {
      int row0 = m0 + wr + tm * 16 + ((lane & 16) ? 8 : 0);
      int col  = n0 + wc + tn * 16 + (lane & 15);
      int seg = col >> 9, cl = col & 511;
#pragma unroll
      for (int r = 0; r < 8; ++r) {
        float v = acc[tm][tn][r];
        size_t idx = (size_t)(row0 + r) * HD_ + cl;
        if (seg == 0)      qh[idx] = (_Float16)v;
        else if (seg == 1) kh[idx] = (_Float16)v;
        else if (seg == 2) vh[idx] = (_Float16)v;
        else               oh[idx] = (_Float16)(1.0f / (1.0f + expf(-(v + bo[cl]))));
      }
    }
}

// ---------------------------------------------------------------------------
// K2: exponential gates  ig/fg[b,h,t] = exp(x . W + b). One wave per dot.
// ---------------------------------------------------------------------------
__global__ __launch_bounds__(512) void k_gates(
    const float* __restrict__ x, const float* __restrict__ Wi,
    const float* __restrict__ bi, const float* __restrict__ Wf,
    const float* __restrict__ bf, float* __restrict__ ig,
    float* __restrict__ fg) {
  int m = blockIdx.x;
  int b = m >> 11, t = m & 2047;
  int tid = threadIdx.x, lane = tid & 31, w = tid >> 5;  // 16 waves
  int head = w >> 1, gate = w & 1;
  const float* W  = gate ? Wf : Wi;
  const float* xr = x + (size_t)m * IN_;
  const float* wr = W + (size_t)head * IN_;
  float s = 0.f;
  for (int k = lane; k < IN_; k += 32) s += xr[k] * wr[k];
#pragma unroll
  for (int off = 16; off > 0; off >>= 1) s += __shfl_xor(s, off, 32);
  if (lane == 0) {
    float g = expf(s + (gate ? bf[head] : bi[head]));
    size_t idx = ((size_t)(b * H_ + head)) * T_ + t;
    if (gate) fg[idx] = g; else ig[idx] = g;
  }
}

// ---------------------------------------------------------------------------
// K3: sequential chunk-state scan (64 blocks = (b,h), 64 threads, thread d
//     owns row d of C in registers). Emits chunk-initial C0/n0, cumulative
//     forget A, and final (C,n) into d_out tail.
// ---------------------------------------------------------------------------
__global__ __launch_bounds__(64) void k_scan(
    const _Float16* __restrict__ kh, const _Float16* __restrict__ vh,
    const float* __restrict__ ig, const float* __restrict__ fg,
    float* __restrict__ Abuf, float* __restrict__ Cinit,
    float* __restrict__ ninit, float* __restrict__ outC,
    float* __restrict__ outn) {
  int blk = blockIdx.x;
  int b = blk >> 3, h = blk & 7;
  int d = threadIdx.x;
  __shared__ float Kc[64][65];
  __shared__ float Vc[64][65];
  __shared__ float fv[64], iv[64], Av[64], coef[64];
  float C[64];
#pragma unroll
  for (int e = 0; e < 64; ++e) C[e] = 0.f;
  float nv = 0.f;
  for (int c = 0; c < NC; ++c) {
    size_t cb = ((size_t)blk * NC + c) * 4096;
    for (int e = 0; e < 64; ++e) Cinit[cb + (size_t)d * 64 + e] = C[e];
    ninit[((size_t)blk * NC + c) * 64 + d] = nv;
    int t0 = c * 64;
    // prefetch next chunk's K/V stream into cache (global_prefetch_b8)
    if (c + 1 < NC) {
      size_t gn = (size_t)(b * T_ + t0 + 64 + (d & 63)) * HD_ + h * 64;
      __builtin_prefetch(kh + gn, 0, 1);
      __builtin_prefetch(vh + gn, 0, 1);
    }
    for (int i = d; i < 4096; i += 64) {
      int s = i >> 6, e = i & 63;
      size_t gi = (size_t)(b * T_ + t0 + s) * HD_ + h * 64 + e;
      Kc[s][e] = (float)kh[gi];
      Vc[s][e] = (float)vh[gi];
    }
    fv[d] = fg[(size_t)blk * T_ + t0 + d];
    iv[d] = ig[(size_t)blk * T_ + t0 + d];
    __syncthreads();
    if (d == 0) {
      float a = 1.f;
      for (int s = 0; s < 64; ++s) { a *= fv[s]; Av[s] = a; }
    }
    __syncthreads();
    float Ft = Av[63];
    coef[d] = iv[d] * Ft / Av[d];
    Abuf[(size_t)blk * T_ + t0 + d] = Av[d];
    __syncthreads();
#pragma unroll
    for (int e = 0; e < 64; ++e) C[e] *= Ft;
    nv *= Ft;
    for (int s = 0; s < 64; ++s) {
      float wv = coef[s] * Vc[s][d];
      nv += coef[s] * Kc[s][d];
#pragma unroll
      for (int e = 0; e < 64; ++e) C[e] += wv * Kc[s][e];
    }
    __syncthreads();
  }
  size_t co = (size_t)blk * 4096;
  for (int e = 0; e < 64; ++e) outC[co + (size_t)d * 64 + e] = C[e];
  outn[(size_t)blk * 64 + d] = nv;
}

// ---------------------------------------------------------------------------
// K4: chunk-parallel retrieval, 2048 blocks = (b,h,c), 8 waves.
//     S = tril(Qhat Khat^T);  H = S@V + Qhat@C0^T;  h = H/denom * o.
// ---------------------------------------------------------------------------
#define LDP 72
__global__ __launch_bounds__(256) void k_chunk(
    const _Float16* __restrict__ qh, const _Float16* __restrict__ kh,
    const _Float16* __restrict__ vh, const _Float16* __restrict__ oh,
    const float* __restrict__ ig, const float* __restrict__ Abuf,
    const float* __restrict__ Cinit, const float* __restrict__ ninit,
    _Float16* __restrict__ hs) {
  int blk = blockIdx.x;
  int c  = blk & (NC - 1);
  int bh = blk >> 5;
  int b = bh >> 3, h = bh & 7;
  int tid = threadIdx.x, lane = tid & 31, w = tid >> 5;
  __shared__ _Float16 Qs[64 * LDP];
  __shared__ _Float16 Ks[64 * LDP];
  __shared__ _Float16 Vs[64 * LDP];
  __shared__ _Float16 Ss[64 * LDP];
  __shared__ _Float16 Cs[64 * LDP];
  __shared__ float Aj[64], iv[64], n0[64], den[64];
  int t0 = c * 64;
  if (tid < 64) {
    Aj[tid] = Abuf[(size_t)bh * T_ + t0 + tid];
    iv[tid] = ig[(size_t)bh * T_ + t0 + tid];
    n0[tid] = ninit[((size_t)bh * NC + c) * 64 + tid];
  }
  __syncthreads();
  size_t cbase = ((size_t)bh * NC + c) * 4096;
  for (int i = tid; i < 4096; i += 256) {
    int r = i >> 6, e = i & 63;
    size_t gi = (size_t)(b * T_ + t0 + r) * HD_ + h * 64 + e;
    Qs[r * LDP + e] = (_Float16)((float)qh[gi] * Aj[r]);
    Ks[r * LDP + e] = (_Float16)((float)kh[gi] * (iv[r] / Aj[r]));
    Vs[r * LDP + e] = vh[gi];
    Cs[r * LDP + e] = (_Float16)Cinit[cbase + i];
  }
  __syncthreads();
  // GEMM1: S = Qhat @ Khat^T  (M=j, N=s, K=d) + causal mask -> Ss
  {
    v8f sacc[2] = {};
#pragma unroll
    for (int t = 0; t < 2; ++t) {
      int ti = 2 * w + t;
      int r0 = (ti >> 2) * 16, c0 = (ti & 3) * 16;
      for (int k0 = 0; k0 < 64; k0 += 32)
        sacc[t] = wmma_f16(frag_A(Qs, LDP, r0, k0, lane),
                           frag_B_nmajor(Ks, LDP, c0, k0, lane), sacc[t]);
#pragma unroll
      for (int r = 0; r < 8; ++r) {
        int j = r0 + ((lane & 16) ? 8 : 0) + r;
        int s = c0 + (lane & 15);
        Ss[j * LDP + s] = (_Float16)((s <= j) ? sacc[t][r] : 0.f);
      }
    }
  }
  __syncthreads();
  if (tid < 64) {
    int j = tid;
    float rs = 0.f, qn = 0.f;
    for (int e = 0; e < 64; ++e) {
      rs += (float)Ss[j * LDP + e];
      qn += (float)Qs[j * LDP + e] * n0[e];
    }
    float dn = fabsf(qn + rs);
    den[j] = dn > 1.f ? dn : 1.f;
  }
  __syncthreads();
  // GEMM2+3 fused: H = Ss@V + Qhat@C0^T  (M=j, N=d)
  {
    v8f hacc[2] = {};
#pragma unroll
    for (int t = 0; t < 2; ++t) {
      int ti = 2 * w + t;
      int r0 = (ti >> 2) * 16, c0 = (ti & 3) * 16;
      for (int k0 = 0; k0 < 64; k0 += 32) {
        hacc[t] = wmma_f16(frag_A(Ss, LDP, r0, k0, lane),
                           frag_B_kmajor(Vs, LDP, k0, c0, lane), hacc[t]);
        hacc[t] = wmma_f16(frag_A(Qs, LDP, r0, k0, lane),
                           frag_B_nmajor(Cs, LDP, c0, k0, lane), hacc[t]);
      }
#pragma unroll
      for (int r = 0; r < 8; ++r) {
        int j = r0 + ((lane & 16) ? 8 : 0) + r;
        int d = c0 + (lane & 15);
        size_t gi = (size_t)(b * T_ + t0 + j) * HD_ + h * 64 + d;
        hs[gi] = (_Float16)(hacc[t][r] / den[j] * (float)oh[gi]);
      }
    }
  }
}

// ---------------------------------------------------------------------------
// K5a: LayerNorm over HD=512 per token -> hn (f16).
// ---------------------------------------------------------------------------
__global__ __launch_bounds__(128) void k_ln(
    const _Float16* __restrict__ hs, const float* __restrict__ lng,
    const float* __restrict__ lnb, _Float16* __restrict__ hn) {
  int m = blockIdx.x, tid = threadIdx.x;
  __shared__ float red[128];
  const _Float16* row = hs + (size_t)m * HD_;
  float v[4];
  float s = 0.f;
#pragma unroll
  for (int i = 0; i < 4; ++i) { v[i] = (float)row[tid + i * 128]; s += v[i]; }
  red[tid] = s; __syncthreads();
  for (int off = 64; off > 0; off >>= 1) {
    if (tid < off) red[tid] += red[tid + off];
    __syncthreads();
  }
  float mu = red[0] * (1.0f / 512.0f);
  __syncthreads();
  float s2 = 0.f;
#pragma unroll
  for (int i = 0; i < 4; ++i) { float d = v[i] - mu; s2 += d * d; }
  red[tid] = s2; __syncthreads();
  for (int off = 64; off > 0; off >>= 1) {
    if (tid < off) red[tid] += red[tid + off];
    __syncthreads();
  }
  float rstd = rsqrtf(red[0] * (1.0f / 512.0f) + 1e-5f);
#pragma unroll
  for (int i = 0; i < 4; ++i) {
    int col = tid + i * 128;
    hn[(size_t)m * HD_ + col] = (_Float16)((v[i] - mu) * rstd * lng[col] + lnb[col]);
  }
}

// ---------------------------------------------------------------------------
// K5b: out = hn @ W_out^T  (16384 x 1024), f32 output, TDM staging.
// ---------------------------------------------------------------------------
__global__ __launch_bounds__(128) void k_out(
    const _Float16* __restrict__ hn, const _Float16* __restrict__ Wouth,
    float* __restrict__ out) {
  __shared__ _Float16 As[2][64 * LDA];
  __shared__ _Float16 Bs[2][64 * LDA];
  int tid = threadIdx.x, lane = tid & 31, w = tid >> 5;
  int m0 = blockIdx.y * 64, n0 = blockIdx.x * 64;
  int wr = (w >> 1) * 32, wc = (w & 1) * 32;
  const _Float16* Ab = hn + (size_t)m0 * HD_;
  const _Float16* Bb = Wouth + (size_t)n0 * HD_;
  v8f acc[2][2] = {};
  stage_pair(Ab, Bb, As[0], Bs[0], HD_, tid, 128);
  stage_wait(tid);
  int p = 0;
  for (int k0 = 0; k0 < HD_; k0 += 32) {
    if (k0 + 32 < HD_)
      stage_pair(Ab + k0 + 32, Bb + k0 + 32, As[p ^ 1], Bs[p ^ 1], HD_, tid, 128);
    v16h a0 = frag_A(As[p], LDA, wr, 0, lane);
    v16h a1 = frag_A(As[p], LDA, wr + 16, 0, lane);
    v16h b0 = frag_B_nmajor(Bs[p], LDA, wc, 0, lane);
    v16h b1 = frag_B_nmajor(Bs[p], LDA, wc + 16, 0, lane);
    acc[0][0] = wmma_f16(a0, b0, acc[0][0]);
    acc[0][1] = wmma_f16(a0, b1, acc[0][1]);
    acc[1][0] = wmma_f16(a1, b0, acc[1][0]);
    acc[1][1] = wmma_f16(a1, b1, acc[1][1]);
    stage_wait(tid);
    p ^= 1;
  }
#pragma unroll
  for (int tm = 0; tm < 2; ++tm)
#pragma unroll
    for (int tn = 0; tn < 2; ++tn) {
      int row0 = m0 + wr + tm * 16 + ((lane & 16) ? 8 : 0);
      int col  = n0 + wc + tn * 16 + (lane & 15);
#pragma unroll
      for (int r = 0; r < 8; ++r)
        out[(size_t)(row0 + r) * HID_ + col] = acc[tm][tn][r];
    }
}

// ---------------------------------------------------------------------------
extern "C" void kernel_launch(void* const* d_in, const int* in_sizes, int n_in,
                              void* d_out, int out_size, void* d_ws,
                              size_t ws_size, hipStream_t stream) {
  const float* x    = (const float*)d_in[0];
  const float* Wq   = (const float*)d_in[1];
  const float* Wk   = (const float*)d_in[2];
  const float* Wv   = (const float*)d_in[3];
  const float* Wi   = (const float*)d_in[4];
  const float* bi   = (const float*)d_in[5];
  const float* Wf   = (const float*)d_in[6];
  const float* bf   = (const float*)d_in[7];
  const float* Wo   = (const float*)d_in[8];
  const float* bo   = (const float*)d_in[9];
  const float* Wout = (const float*)d_in[10];
  const float* lng  = (const float*)d_in[11];
  const float* lnb  = (const float*)d_in[12];

  char* ws = (char*)d_ws;
  size_t off = 0;
  auto carve = [&](size_t bytes) -> char* {
    char* p = ws + off;
    off = (off + bytes + 255) & ~(size_t)255;
    return p;
  };
  _Float16* Wcat  = (_Float16*)carve((size_t)NCAT * IN_ * 2);
  _Float16* Wouth = (_Float16*)carve((size_t)HID_ * HD_ * 2);
  _Float16* xh    = (_Float16*)carve((size_t)NTOK * IN_ * 2);
  _Float16* qh    = (_Float16*)carve((size_t)NTOK * HD_ * 2);
  _Float16* kh    = (_Float16*)carve((size_t)NTOK * HD_ * 2);
  _Float16* vh    = (_Float16*)carve((size_t)NTOK * HD_ * 2);
  _Float16* oh    = (_Float16*)carve((size_t)NTOK * HD_ * 2);
  float*    ig    = (float*)carve((size_t)B_ * H_ * T_ * 4);
  float*    fg    = (float*)carve((size_t)B_ * H_ * T_ * 4);
  float*    Abuf  = (float*)carve((size_t)B_ * H_ * T_ * 4);
  float*    Cinit = (float*)carve((size_t)B_ * H_ * NC * 4096 * 4);
  float*    ninit = (float*)carve((size_t)B_ * H_ * NC * 64 * 4);
  _Float16* hsbuf = (_Float16*)carve((size_t)NTOK * HD_ * 2);
  _Float16* hnbuf = (_Float16*)carve((size_t)NTOK * HD_ * 2);

  float* out  = (float*)d_out;
  float* outC = out + (size_t)NTOK * HID_;
  float* outn = outC + (size_t)B_ * H_ * D_ * D_;

  k_cast<<<(NCAT * IN_ + HID_ * HD_ + 255) / 256, 256, 0, stream>>>(
      Wq, Wk, Wv, Wo, Wout, Wcat, Wouth);
  k_castx<<<(NTOK * IN_) / 1024, 256, 0, stream>>>(x, xh);
  k_proj<<<dim3(NCAT / 64, NTOK / 64), 128, 0, stream>>>(
      xh, Wcat, bo, qh, kh, vh, oh);
  k_gates<<<NTOK, 512, 0, stream>>>(x, Wi, bi, Wf, bf, ig, fg);
  k_scan<<<B_ * H_, 64, 0, stream>>>(kh, vh, ig, fg, Abuf, Cinit, ninit,
                                     outC, outn);
  k_chunk<<<B_ * H_ * NC, 256, 0, stream>>>(qh, kh, vh, oh, ig, Abuf, Cinit,
                                            ninit, hsbuf);
  k_ln<<<NTOK, 128, 0, stream>>>(hsbuf, lng, lnb, hnbuf);
  k_out<<<dim3(HID_ / 64, NTOK / 64), 128, 0, stream>>>(hnbuf, Wouth, out);
}
